// LieNet_33749853012668
// MI455X (gfx1250) — matrix operand order repaired and checked by reading, not compiled, not run on hardware
//
#include <hip/hip_runtime.h>
#include <hip/hip_bf16.h>
#include <hip/hip_fp16.h>

typedef __attribute__((ext_vector_type(16))) _Float16 v16h;
typedef __attribute__((ext_vector_type(8)))  _Float16 v8h;
typedef __attribute__((ext_vector_type(4)))  _Float16 v4h;
typedef __attribute__((ext_vector_type(8)))  float    v8f;
typedef __attribute__((ext_vector_type(4)))  float    v4f;

#define SLEN 2048
#define BATCH 32
#define DMODEL 1024
#define NHEAD 16
#define HDIM 64
#define K2 128            // 2*HDIM, GEMM K per head
#define EPSV 1e-5f

// LDS row stride: 2056 halves = 4112 bytes = 16*257 -> consecutive rows land on
// different bank groups for 16-byte ds loads. Same byte stride reused as 1028 f32.
#define ROWH 2056
#define ROWF 1028

// Branch-free GELU (tanh form): gelu(x) = x * sigmoid(2u), u = sqrt(2/pi)(x + 0.044715 x^3)
// sigmoid via v_exp_f32 + v_rcp_f32 -> ~6 straight-line VALU ops, no EXEC divergence.
__device__ __forceinline__ float gelu_fast(float x) {
    const float u2 = 1.5957691216057308f * x * __builtin_fmaf(0.044715f * x, x, 1.0f); // 2u
    const float e  = __builtin_amdgcn_exp2f(u2 * 1.4426950408889634f);                 // e^{2u}
    const float r  = __builtin_amdgcn_rcpf(e + 1.0f);                                  // 1/(1+e^{2u})
    return x - x * r;                                                                  // x*(1-r)
}

// ---------------------------------------------------------------------------
// Pack W [H,128,64] f32 into per-tile f16 B fragments, lane-consumable layout:
//   tile = (h, kt, nt); lane 0-15 -> column n = nt*16+lane,      K = kt*32 + 0..15
//                       lane 16-31 -> column n = nt*16+(lane-16), K = kt*32 + 16..31
// 16 halves per lane, contiguous (two b128 stores / loads).
// ---------------------------------------------------------------------------
__global__ void wpack_kernel(const float* __restrict__ W, _Float16* __restrict__ wsW) {
    const int tile = blockIdx.x;       // h*16 + kt*4 + nt, 256 tiles
    const int lane = threadIdx.x;      // 0..31
    const int h  = tile >> 4;
    const int kt = (tile >> 2) & 3;
    const int nt = tile & 3;
    const int n = nt * 16 + (lane & 15);
    const int kbase = kt * 32 + ((lane < 16) ? 0 : 16);
    const float* Wh = W + (size_t)h * (K2 * HDIM);
    v16h frag;
#pragma unroll
    for (int j = 0; j < 16; ++j)
        frag[j] = (_Float16)Wh[(size_t)(kbase + j) * HDIM + n];
    *(v16h*)(wsW + ((size_t)tile * 32 + lane) * 16) = frag;
}

// ---------------------------------------------------------------------------
// One workgroup (8 waves / 256 threads) per timestep t.
// ---------------------------------------------------------------------------
__global__ void __launch_bounds__(256)
lienet_step_kernel(const float* __restrict__ src,
                   const _Float16* __restrict__ wsW,
                   const float* __restrict__ bias,
                   const float* __restrict__ gamma,
                   const float* __restrict__ beta,
                   float* __restrict__ out) {
    extern __shared__ char lds_raw[];
    _Float16* lds = (_Float16*)lds_raw;      // phase1/2: [32][ROWH] f16 cat layout
    float*    yld = (float*)lds_raw;         // phase2/3: [32][ROWF] f32 activations

    const int t    = blockIdx.x;
    const int tid  = threadIdx.x;
    const int wave = tid >> 5;
    const int lane = tid & 31;

    const float* xbase = src + (size_t)t * BATCH * DMODEL;
    const float* cbase = src + (size_t)(t > 0 ? t - 1 : 0) * BATCH * DMODEL;

    // ---------------- Phase 1: build A = [LN(src[t-1]) | src[t]] in f16 -----
    for (int rr = 0; rr < 4; ++rr) {
        const int row = wave * 4 + rr;

        // x half: straight f32 -> f16 convert
        {
            const float* xr = xbase + (size_t)row * DMODEL;
#pragma unroll
            for (int i = 0; i < 8; ++i) {
                const int d = i * 128 + lane * 4;        // coalesced float4
                v4f v = *(const v4f*)(xr + d);
                const int hd = d >> 6, k = d & 63;
                v4h p;
                p[0] = (_Float16)v[0]; p[1] = (_Float16)v[1];
                p[2] = (_Float16)v[2]; p[3] = (_Float16)v[3];
                *(v4h*)(lds + (size_t)row * ROWH + hd * K2 + 64 + k) = p;
            }
        }
        // c half: LayerNorm of previous timestep (zeros at t==0)
        if (t == 0) {
            const _Float16 z = (_Float16)0.0f;
            v4h p; p[0] = z; p[1] = z; p[2] = z; p[3] = z;
#pragma unroll
            for (int i = 0; i < 8; ++i) {
                const int d = i * 128 + lane * 4;
                const int hd = d >> 6, k = d & 63;
                *(v4h*)(lds + (size_t)row * ROWH + hd * K2 + k) = p;
            }
        } else {
            const float* cr = cbase + (size_t)row * DMODEL;
            v4f vals[8];
            float s = 0.f, s2 = 0.f;
#pragma unroll
            for (int i = 0; i < 8; ++i) {
                const int d = i * 128 + lane * 4;
                vals[i] = *(const v4f*)(cr + d);
#pragma unroll
                for (int j = 0; j < 4; ++j) { s += vals[i][j]; s2 += vals[i][j] * vals[i][j]; }
            }
#pragma unroll
            for (int off = 16; off > 0; off >>= 1) {
                s  += __shfl_xor(s,  off, 32);
                s2 += __shfl_xor(s2, off, 32);
            }
            const float mu = s * (1.0f / (float)DMODEL);
            const float rs = rsqrtf(s2 * (1.0f / (float)DMODEL) - mu * mu + EPSV);
#pragma unroll
            for (int i = 0; i < 8; ++i) {
                const int d = i * 128 + lane * 4;
                const int hd = d >> 6, k = d & 63;
                v4f g  = *(const v4f*)(gamma + d);
                v4f be = *(const v4f*)(beta + d);
                v4h p;
#pragma unroll
                for (int j = 0; j < 4; ++j)
                    p[j] = (_Float16)((vals[i][j] - mu) * rs * g[j] + be[j]);
                *(v4h*)(lds + (size_t)row * ROWH + hd * K2 + k) = p;
            }
        }
    }
    __syncthreads();

    // ---------------- Phase 2: per-head GEMM via WMMA + bias + GELU ---------
    // Wave owns heads h0, h0+1.  A fragment (contiguous K order, same as packed B):
    // lanes 0-15: row M = lane,  K runs [kt*32+0,16) contiguous
    // lanes16-31: row M = lane-16, K runs [kt*32+16,32) contiguous.
    const int h0 = wave * 2;
    v16h afr[2][2][4];                         // [head][mt][kt]
#pragma unroll
    for (int hh = 0; hh < 2; ++hh) {
        const int h = h0 + hh;
#pragma unroll
        for (int mt = 0; mt < 2; ++mt) {
            const int row = mt * 16 + (lane & 15);
            const _Float16* abase = lds + (size_t)row * ROWH + h * K2;
#pragma unroll
            for (int kt = 0; kt < 4; ++kt) {
                const int kb = kt * 32 + ((lane < 16) ? 0 : 16);
                v8h lo = *(const v8h*)(abase + kb);        // ds_load_b128
                v8h hi = *(const v8h*)(abase + kb + 8);    // ds_load_b128
                v16h f;
#pragma unroll
                for (int j = 0; j < 8; ++j) { f[j] = lo[j]; f[8 + j] = hi[j]; }
                afr[hh][mt][kt] = f;
            }
        }
    }
    __syncthreads();   // everyone done reading f16 A before y overwrites LDS

#pragma unroll
    for (int hh = 0; hh < 2; ++hh) {
        const int h = h0 + hh;
#pragma unroll
        for (int nt = 0; nt < 4; ++nt) {
            // Prefetch all 4 K-fragments of this (head, nt) tile in one clause,
            // so the 8 WMMAs below run with graduated loadcnt waits.
            v16h bf[4];
#pragma unroll
            for (int kt = 0; kt < 4; ++kt)
                bf[kt] = *(const v16h*)(wsW + (((size_t)(h * 4 + kt) * 4 + nt) * 32 + lane) * 16);

            v8f acc0 = {}, acc1 = {};
#pragma unroll
            for (int kt = 0; kt < 4; ++kt) {
                acc0 = __builtin_amdgcn_wmma_f32_16x16x32_f16(
                    false, afr[hh][0][kt], false, bf[kt], (short)0, acc0, false, false);
                acc1 = __builtin_amdgcn_wmma_f32_16x16x32_f16(
                    false, afr[hh][1][kt], false, bf[kt], (short)0, acc1, false, false);
            }
            const int ncol = nt * 16 + (lane & 15);
            const int colg = h * HDIM + ncol;
            const float bb = bias[colg];
            // C/D layout: VGPR r, lanes 0-15 -> M=r, lanes 16-31 -> M=r+8
#pragma unroll
            for (int r = 0; r < 8; ++r) {
                const int row0 = r + ((lane < 16) ? 0 : 8);
                yld[(size_t)row0 * ROWF + colg]        = gelu_fast(acc0[r] + bb);
                yld[(size_t)(row0 + 16) * ROWF + colg] = gelu_fast(acc1[r] + bb);
            }
        }
    }
    __syncthreads();

    // ---------------- Phase 3: LayerNorm(y) per batch row, write out --------
    for (int rr = 0; rr < 4; ++rr) {
        const int row = wave * 4 + rr;
        const float* yr = yld + (size_t)row * ROWF;
        float s = 0.f, s2 = 0.f;
#pragma unroll
        for (int i = 0; i < 8; ++i) {
            const int d = i * 128 + lane * 4;
            v4f v = *(const v4f*)(yr + d);
#pragma unroll
            for (int j = 0; j < 4; ++j) { s += v[j]; s2 += v[j] * v[j]; }
        }
#pragma unroll
        for (int off = 16; off > 0; off >>= 1) {
            s  += __shfl_xor(s,  off, 32);
            s2 += __shfl_xor(s2, off, 32);
        }
        const float mu = s * (1.0f / (float)DMODEL);
        const float rs = rsqrtf(s2 * (1.0f / (float)DMODEL) - mu * mu + EPSV);
        float* orow = out + ((size_t)t * BATCH + row) * DMODEL;
#pragma unroll
        for (int i = 0; i < 8; ++i) {
            const int d = i * 128 + lane * 4;
            v4f v  = *(const v4f*)(yr + d);
            v4f g  = *(const v4f*)(gamma + d);
            v4f be = *(const v4f*)(beta + d);
            v4f o;
#pragma unroll
            for (int j = 0; j < 4; ++j)
                o[j] = (v[j] - mu) * rs * g[j] + be[j];
            *(v4f*)(orow + d) = o;                         // global_store_b128
        }
    }
}

extern "C" void kernel_launch(void* const* d_in, const int* in_sizes, int n_in,
                              void* d_out, int out_size, void* d_ws, size_t ws_size,
                              hipStream_t stream) {
    const float* src   = (const float*)d_in[0];   // [S,B,D]
    const float* W     = (const float*)d_in[1];   // [H,2*DIM,DIM]
    const float* bias  = (const float*)d_in[2];   // [H,DIM]
    const float* gamma = (const float*)d_in[3];   // [D]
    const float* beta  = (const float*)d_in[4];   // [D]
    float* out = (float*)d_out;                   // [S,B,D]
    _Float16* wsW = (_Float16*)d_ws;              // 256 KB packed f16 W fragments

    // Pack W into WMMA B fragments (tiny; L2-resident thereafter).
    wpack_kernel<<<NHEAD * 4 * 4, 32, 0, stream>>>(W, wsW);

    // One workgroup per timestep; 128.5 KB dynamic LDS (gfx1250 WGP has 320 KB).
    const size_t ldsBytes = (size_t)BATCH * ROWH * sizeof(_Float16);   // 131584
    lienet_step_kernel<<<SLEN, 256, ldsBytes, stream>>>(src, wsW, bias, gamma, beta, out);
}
